// Transducer_36618891165721
// MI455X (gfx1250) — matrix operand order repaired and checked by reading, not compiled, not run on hardware
//
#include <hip/hip_runtime.h>
#include <hip/hip_bf16.h>

// ---------------------------------------------------------------------------
// Transducer joint network for MI455X (gfx1250, wave32, WMMA).
//   out[b,u,t,v] = tanh(p[b,u,:] + t[b,t,:] + b1) . W2[:,v] + b2[v]
//   p = predict @ W1[:512], t = transcript @ W1[512:]
// Big GEMM: M=48000, N=4096, K=512 -> v_wmma_f32_16x16x32_f16, 2x2 wave tile.
// ---------------------------------------------------------------------------

#define B_  2
#define T_  300
#define U_  80
#define K_  512      // HID == H_T == H_P
#define V_  4096

typedef _Float16 v16h __attribute__((ext_vector_type(16)));
typedef _Float16 v8h  __attribute__((ext_vector_type(8)));
typedef float    v8f  __attribute__((ext_vector_type(8)));
typedef float    v4f  __attribute__((ext_vector_type(4)));

union Frag16 { v16h v; v8h h8[2]; };

// ---------------------------------------------------------------------------
// Kernel 1: p = predict @ W1[:512], t = transcript @ W1[512:]  (fp32, tiny)
// ---------------------------------------------------------------------------
__global__ __launch_bounds__(256) void proj_kernel(
    const float* __restrict__ transcript,   // (B,T,512)
    const float* __restrict__ predict,      // (B,U,512)
    const float* __restrict__ w1,           // (1024,512)
    float* __restrict__ pOut,               // (B,U,512)
    float* __restrict__ tOut)               // (B,T,512)
{
    __shared__ float xs[K_];
    const int r   = blockIdx.x;             // 0 .. B*U + B*T - 1
    const int tid = threadIdx.x;

    const float* x; const float* W; float* o;
    if (r < B_ * U_) {
        x = predict + (size_t)r * K_;
        W = w1;                              // Wp = W1[:512]
        o = pOut + (size_t)r * K_;
    } else {
        const int rr = r - B_ * U_;
        x = transcript + (size_t)rr * K_;
        W = w1 + (size_t)K_ * K_;            // Wt = W1[512:]
        o = tOut + (size_t)rr * K_;
    }

    xs[tid]       = x[tid];
    xs[tid + 256] = x[tid + 256];
    __syncthreads();

    float acc0 = 0.f, acc1 = 0.f;
    for (int k = 0; k < K_; ++k) {
        const float xv = xs[k];
        acc0 = fmaf(xv, W[(size_t)k * K_ + tid],        acc0);
        acc1 = fmaf(xv, W[(size_t)k * K_ + tid + 256],  acc1);
    }
    o[tid]       = acc0;
    o[tid + 256] = acc1;
}

// ---------------------------------------------------------------------------
// Kernel 2: W2 (512 x 4096 fp32) -> W2t (4096 x 512 f16), K-contiguous rows
// so each WMMA B-fragment is two contiguous global_load_b128 per lane.
// ---------------------------------------------------------------------------
__global__ __launch_bounds__(256) void w2_transpose_kernel(
    const float* __restrict__ w2,           // (512,4096)
    _Float16* __restrict__ w2t)             // (4096,512)
{
    __shared__ float tile[32][33];
    const int v0 = blockIdx.x * 32;         // 128 blocks over V
    const int k0 = blockIdx.y * 32;         // 16  blocks over K
    const int tx = threadIdx.x & 31;
    const int ty = threadIdx.x >> 5;        // 0..7

    #pragma unroll
    for (int i = 0; i < 4; ++i) {
        const int k = k0 + ty + i * 8;
        tile[ty + i * 8][tx] = w2[(size_t)k * V_ + v0 + tx];
    }
    __syncthreads();
    #pragma unroll
    for (int i = 0; i < 4; ++i) {
        const int v = v0 + ty + i * 8;
        w2t[(size_t)v * K_ + k0 + tx] = (_Float16)tile[tx][ty + i * 8];
    }
}

// ---------------------------------------------------------------------------
// Kernel 3: fused tanh + GEMM via v_wmma_f32_16x16x32_f16, 2x2 wave tiling.
// Block = 32 (u,t) pairs. Stage h = tanh(p+t+b1) once to LDS (f16, padded),
// then 8 waves sweep the 128 N-pairs; each wave holds 4 accumulators
// (2 M-tiles x 2 N-tiles) so per WMMA traffic is 1 ds_b128 + 1 global_b128.
// ---------------------------------------------------------------------------
#define HPITCH (K_ + 8)   // halves; 16B row pad to spread LDS banks
#define MROWS  32         // M rows staged per block

__global__ __launch_bounds__(256) void joint_kernel(
    const float*    __restrict__ p,         // (B,U,512) fp32
    const float*    __restrict__ t,         // (B,T,512) fp32
    const float*    __restrict__ b1,        // (512)
    const _Float16* __restrict__ w2t,       // (4096,512) f16
    const float*    __restrict__ b2,        // (4096)
    float*          __restrict__ out)       // (B,U,T,4096)
{
    __shared__ _Float16 hTile[MROWS * HPITCH];

    const int tid   = threadIdx.x;
    const int blk   = blockIdx.x;           // 0..1499
    const int b     = blk / 750;
    const int mtile = blk % 750;
    const int m0    = mtile * MROWS;        // U_*T_ = 24000 = 750*32

    // ---- stage h tile: 32 rows x 512 K, f16, tanh fused here (once) ----
    {
        const int row = tid >> 3;           // 0..31
        const int sub = tid & 7;            // 0..7  -> 64 K each
        const int m   = m0 + row;
        const int u   = m / T_;
        const int tt  = m - u * T_;
        const float* prow = p + ((size_t)b * U_ + u)  * K_;
        const float* trow = t + ((size_t)b * T_ + tt) * K_;
        const int k0 = sub * 64;
        #pragma unroll
        for (int c = 0; c < 16; ++c) {
            const int k = k0 + c * 4;
            const v4f pv = *(const v4f*)(prow + k);
            const v4f tv = *(const v4f*)(trow + k);
            const v4f bv = *(const v4f*)(b1   + k);
            _Float16* dst = &hTile[row * HPITCH + k];
            #pragma unroll
            for (int j = 0; j < 4; ++j)
                dst[j] = (_Float16)tanhf(pv[j] + tv[j] + bv[j]);
        }
    }
    __syncthreads();

    // ---- WMMA sweep over N, 2 M-tiles x 2 N-tiles per wave ----
    const int wave = tid >> 5;              // 0..7
    const int lane = tid & 31;
    const int lrow = lane & 15;
    const int lgrp = lane >> 4;             // 0 / 1

    // A fragments (16x32 f16, ISA layout): lane group g reads, per K-step,
    // halves K = 32*kb + 8g .. +7 and K = 32*kb + 16 + 8g .. +7 of its row.
    const _Float16* a0Base = &hTile[(lrow     ) * HPITCH + lgrp * 8]; // rows 0..15
    const _Float16* a1Base = &hTile[(lrow + 16) * HPITCH + lgrp * 8]; // rows 16..31

    // Output row bases for this lane's C rows (M = r + 8*lgrp [+16]).
    size_t rowBaseA[8], rowBaseB[8];
    #pragma unroll
    for (int r = 0; r < 8; ++r) {
        int m  = m0 + r + lgrp * 8;
        int u  = m / T_;
        int tt = m - u * T_;
        rowBaseA[r] = (((size_t)b * U_ + u) * T_ + tt) * (size_t)V_;
        m  += 16;
        u   = m / T_;
        tt  = m - u * T_;
        rowBaseB[r] = (((size_t)b * U_ + u) * T_ + tt) * (size_t)V_;
    }

    for (int it = 0; it < 16; ++it) {       // 8 waves * 16 iters * 2 = 256 N-tiles
        const int n0 = (it * 8 + wave) * 32;

        // B fragments (32x16 f16, ISA layout): lane reads column n0+lrow,
        // halves K = 32*kb + 8*lgrp .. +7 and K = 32*kb + 16 + 8*lgrp .. +7.
        const _Float16* b0p = w2t + (size_t)(n0      + lrow) * K_ + lgrp * 8;
        const _Float16* b1p = w2t + (size_t)(n0 + 16 + lrow) * K_ + lgrp * 8;

        if (it + 1 < 16) {                  // warm L2/WGP$ for next B rows
            const int n0n = ((it + 1) * 8 + wave) * 32;
            __builtin_prefetch(w2t + (size_t)(n0n      + lrow) * K_, 0, 3);
            __builtin_prefetch(w2t + (size_t)(n0n + 16 + lrow) * K_, 0, 3);
        }

        v8f c00 = {}; v8f c01 = {}; v8f c10 = {}; v8f c11 = {};
        #pragma unroll 2
        for (int kb = 0; kb < 16; ++kb) {
            Frag16 a0, a1, f0, f1;
            const _Float16* ap0 = a0Base + kb * 32;
            const _Float16* ap1 = a1Base + kb * 32;
            a0.h8[0] = *(const v8h*)(ap0);          // K = 8g .. 8g+7
            a0.h8[1] = *(const v8h*)(ap0 + 16);     // K = 16+8g .. 16+8g+7
            a1.h8[0] = *(const v8h*)(ap1);
            a1.h8[1] = *(const v8h*)(ap1 + 16);
            f0.h8[0] = *(const v8h*)(b0p + kb * 32);
            f0.h8[1] = *(const v8h*)(b0p + kb * 32 + 16);
            f1.h8[0] = *(const v8h*)(b1p + kb * 32);
            f1.h8[1] = *(const v8h*)(b1p + kb * 32 + 16);

            c00 = __builtin_amdgcn_wmma_f32_16x16x32_f16(
                    false, a0.v, false, f0.v, (short)0, c00, false, false);
            c01 = __builtin_amdgcn_wmma_f32_16x16x32_f16(
                    false, a0.v, false, f1.v, (short)0, c01, false, false);
            c10 = __builtin_amdgcn_wmma_f32_16x16x32_f16(
                    false, a1.v, false, f0.v, (short)0, c10, false, false);
            c11 = __builtin_amdgcn_wmma_f32_16x16x32_f16(
                    false, a1.v, false, f1.v, (short)0, c11, false, false);
        }

        // C/D layout: VGPR r -> M = r + 8*lgrp, N = lrow.
        const int v0 = n0 + lrow;
        const float bias0 = b2[v0];
        const float bias1 = b2[v0 + 16];
        #pragma unroll
        for (int r = 0; r < 8; ++r) {
            out[rowBaseA[r] + v0]      = c00[r] + bias0;
            out[rowBaseA[r] + v0 + 16] = c01[r] + bias1;
            out[rowBaseB[r] + v0]      = c10[r] + bias0;
            out[rowBaseB[r] + v0 + 16] = c11[r] + bias1;
        }
    }
}

// ---------------------------------------------------------------------------
// Launch: proj + transpose (independent) then fused joint, all on `stream`.
// Workspace: p (327,680 B) | t (1,228,800 B) | W2t f16 (4,194,304 B) ~ 5.5 MB
// ---------------------------------------------------------------------------
extern "C" void kernel_launch(void* const* d_in, const int* in_sizes, int n_in,
                              void* d_out, int out_size, void* d_ws, size_t ws_size,
                              hipStream_t stream) {
    const float* transcript = (const float*)d_in[0];
    const float* predict    = (const float*)d_in[1];
    const float* W1         = (const float*)d_in[2];
    const float* b1         = (const float*)d_in[3];
    const float* W2         = (const float*)d_in[4];
    const float* b2         = (const float*)d_in[5];
    float* out = (float*)d_out;

    float*    pBuf = (float*)d_ws;                       // B*U*512
    float*    tBuf = pBuf + (size_t)B_ * U_ * K_;        // B*T*512
    _Float16* w2t  = (_Float16*)(tBuf + (size_t)B_ * T_ * K_);  // V*512 f16

    proj_kernel<<<B_ * U_ + B_ * T_, 256, 0, stream>>>(transcript, predict, W1, pBuf, tBuf);

    dim3 tgrid(V_ / 32, K_ / 32);
    w2_transpose_kernel<<<tgrid, 256, 0, stream>>>(W2, w2t);

    joint_kernel<<<B_ * (U_ * T_ / MROWS), 256, 0, stream>>>(pBuf, tBuf, b1, w2t, b2, out);
}
